// DRNet3_438086664875
// MI455X (gfx1250) — matrix operand order repaired and checked by reading, not compiled, not run on hardware
//
#include <hip/hip_runtime.h>
#include <hip/hip_bf16.h>
#include <math.h>

typedef __attribute__((ext_vector_type(16))) _Float16 v16h;
typedef __attribute__((ext_vector_type(8)))  float    v8f;
typedef int v4i __attribute__((vector_size(16)));

#define AC_LIN   0
#define AC_LRELU 1
#define AC_TANH  2

// CDNA5 async global->LDS copy path (guarded: falls back to direct loads if
// the toolchain does not declare the builtins; host pass is always false).
// Builtin signature (from compiler diagnostics): (v4i AS1*, v4i AS3*, imm, imm).
#if defined(__HIP_DEVICE_COMPILE__) && __has_builtin(__builtin_amdgcn_global_load_async_to_lds_b128) && __has_builtin(__builtin_amdgcn_s_wait_asynccnt)
#define HAVE_ASYNC_LDS 1
#define ASYNC_CP(g, l)                                                        \
  __builtin_amdgcn_global_load_async_to_lds_b128(                             \
      (__attribute__((address_space(1))) v4i*)(g),                            \
      (__attribute__((address_space(3))) v4i*)(l), 0, 0)
#else
#define HAVE_ASYNC_LDS 0
#endif

#define EWIDX size_t i = (size_t)blockIdx.x * blockDim.x + threadIdx.x; if (i >= total) return;

__device__ __forceinline__ float lrelu_f(float v) { return v > 0.f ? v : 0.01f * v; }
__device__ __forceinline__ float clip01(float v) { return fminf(fmaxf(v, 0.f), 1.f); }

// ---------------------------------------------------------------------------
// Weight pre-pack: f32 (Cout,Cin,kh,kw) -> f16 WMMA B-fragment layout
// [coTile][kchunk][lane(32)][e(16)]; K = ci*taps + tap; lanes 0-15 hold
// K=chunk*32+0..15 (N=lane), lanes 16-31 hold K=chunk*32+16..31.
// Out-of-range co/K slots are zero-filled so the conv kernel needs no masks.
// ---------------------------------------------------------------------------
__global__ void pack_w_k(const float* __restrict__ wgt, _Float16* __restrict__ wp,
                         int Cout, int Cin, int tpc, int nCh, size_t total) {
  EWIDX;
  int e = (int)(i & 15); int lane = (int)((i >> 4) & 31); size_t rest = i >> 9;
  int c = (int)(rest % (size_t)nCh); int coT = (int)(rest / (size_t)nCh);
  int hi = lane >> 4; int N = lane & 15;
  int k  = (c << 5) + (hi << 4) + e;
  int co = (coT << 4) + N;
  float v = 0.f;
  int Kreal = Cin * tpc;
  if (k < Kreal && co < Cout) {
    int ci = k / tpc; int tap = k - ci * tpc;
    v = wgt[((size_t)co * Cin + ci) * tpc + tap];
  }
  wp[i] = (_Float16)v;
}

// ---------------------------------------------------------------------------
// Implicit-GEMM conv on v_wmma_f32_16x16x32_f16.
// One wave per block: 16 output pixels (along W) x 32 output channels
// (two accumulators share one A fragment -> 2 WMMA per fragment build).
// LDS: [0..17] zero pad | im2col tile (f16) | u16 K->offset table |
//      (16B-aligned) 2x2KB async double buffer for packed B fragments.
// A-fragment build is two dependent ds_load_u16 per element, no div/mod.
// B fragments for chunk c+1 are staged with GLOBAL_LOAD_ASYNC_TO_LDS_B128
// while chunk c's WMMAs execute (s_wait_asynccnt pipelining); fallback is
// a direct per-lane v16h global load from the packed weight buffer.
// ---------------------------------------------------------------------------
__global__ __launch_bounds__(32) void conv_wmma_k(
    const float* __restrict__ x, const v16h* __restrict__ wp,
    const float* __restrict__ addsrc, float* __restrict__ y,
    int B, int Cin, int H, int W, int Cout, int taps3, int act, int nCh)
{
  extern __shared__ _Float16 sIn[];
  const int lane  = threadIdx.x;
  const int nPair = (Cout + 31) >> 5;
  const int b     = blockIdx.z / nPair;
  const int pair  = blockIdx.z % nPair;
  const int co0   = pair << 5;
  const int hrow  = blockIdx.y;
  const int x0    = blockIdx.x << 4;
  const int tileN = taps3 ? Cin * 54 : Cin << 4;   // halves of im2col tile
  const int Kreal = taps3 ? Cin * 9 : Cin;
  const int Kpad  = nCh << 5;
  unsigned short* sTab = (unsigned short*)(sIn + 18 + tileN);

  if (lane < 18) sIn[lane] = (_Float16)0.f;        // zero pad for dead K slots
  if (taps3) {
    for (int i = lane; i < tileN; i += 32) {       // 3 rows x 18 cols per chan
      int cc = i % 18; int t = i / 18; int r = t % 3; int ci = t / 3;
      int gx = x0 - 1 + cc, gy = hrow - 1 + r;
      float v = 0.f;
      if ((unsigned)gx < (unsigned)W && (unsigned)gy < (unsigned)H)
        v = x[(((size_t)b * Cin + ci) * H + gy) * W + gx];
      sIn[18 + i] = (_Float16)v;
    }
  } else {
    for (int i = lane; i < tileN; i += 32) {
      int cc = i & 15; int ci = i >> 4;
      sIn[18 + i] = (_Float16)x[(((size_t)b * Cin + ci) * H + hrow) * W + x0 + cc];
    }
  }
  // K -> LDS-offset table (im2col decode done once per block, not per chunk)
  for (int k = lane; k < Kpad; k += 32) {
    unsigned short ofs = 0;                        // dead slots -> zero pad
    if (k < Kreal) {
      if (taps3) {
        int ci = k / 9; int tap = k - ci * 9; int ry = tap / 3; int rx = tap - ry * 3;
        ofs = (unsigned short)(18 + ci * 54 + ry * 18 + rx);
      } else {
        ofs = (unsigned short)(18 + (k << 4));
      }
    }
    sTab[k] = ofs;
  }
  if (lane == 0 && hrow + 2 < H)                   // pull next image row to L2
    __builtin_prefetch(&x[(((size_t)b * Cin) * H + (hrow + 2)) * W + x0], 0, 1);
  __syncthreads();

  const int mrow = lane & 15;
  const int hi   = lane >> 4;
  const v16h* wp0 = wp + (size_t)(2 * pair) * nCh * 32 + lane;
  const v16h* wp1 = wp0 + (size_t)nCh * 32;
#if HAVE_ASYNC_LDS
  // 16B-aligned staging region: [buf(2)][tile(2)][lane(32)][32B]
  char* sB = (char*)sIn + ((((size_t)(18 + tileN + Kpad) * 2) + 15) & ~(size_t)15);
  {
    char* l = sB + lane * 32;
    const char* g0 = (const char*)wp0;
    const char* g1 = (const char*)wp1;
    ASYNC_CP(g0,      l);
    ASYNC_CP(g0 + 16, l + 16);
    ASYNC_CP(g1,      l + 1024);
    ASYNC_CP(g1 + 16, l + 1024 + 16);
  }
#endif
  v8f acc0 = {}, acc1 = {};
  for (int c = 0; c < nCh; ++c) {
    const int kb = c << 5;
    v16h a;
#pragma unroll
    for (int e = 0; e < 16; ++e) {
      // A frag (16x32 f16): lanes 0-15 K=0..15, lanes 16-31 K=16..31
      int kA = kb + (hi << 4) + e;
      a[e] = sIn[sTab[kA] + mrow];
    }
    v16h b0, b1;
#if HAVE_ASYNC_LDS
    const int buf = c & 1;
    if (c + 1 < nCh) {                             // stage next chunk, keep 4 in flight
      char* l = sB + (((c + 1) & 1) * 2048) + lane * 32;
      const char* g0 = (const char*)(wp0 + (size_t)(c + 1) * 32);
      const char* g1 = (const char*)(wp1 + (size_t)(c + 1) * 32);
      ASYNC_CP(g0,      l);
      ASYNC_CP(g0 + 16, l + 16);
      ASYNC_CP(g1,      l + 1024);
      ASYNC_CP(g1 + 16, l + 1024 + 16);
      __builtin_amdgcn_s_wait_asynccnt(4);
    } else {
      __builtin_amdgcn_s_wait_asynccnt(0);
    }
    b0 = *(const v16h*)(sB + buf * 2048 + lane * 32);
    b1 = *(const v16h*)(sB + buf * 2048 + 1024 + lane * 32);
#else
    b0 = wp0[c * 32];
    b1 = wp1[c * 32];
#endif
    acc0 = __builtin_amdgcn_wmma_f32_16x16x32_f16(false, a, false, b0, (short)0, acc0,
                                                  false, false);
    acc1 = __builtin_amdgcn_wmma_f32_16x16x32_f16(false, a, false, b1, (short)0, acc1,
                                                  false, false);
  }

  // C/D layout: VGPR r -> M=r (lanes 0-15) / M=r+8 (lanes 16-31); N = lane&15
#pragma unroll
  for (int t = 0; t < 2; ++t) {
    const int ch = co0 + (t << 4) + mrow;
    if (ch < Cout) {
      v8f acc = t ? acc1 : acc0;
#pragma unroll
      for (int r = 0; r < 8; ++r) {
        int px = x0 + r + (hi << 3);
        size_t oi = (((size_t)b * Cout + ch) * H + hrow) * W + px;
        float v = acc[r];
        if (addsrc) v += addsrc[oi];
        if (act == AC_LRELU) v = lrelu_f(v);
        else if (act == AC_TANH) v = tanhf(v);
        y[oi] = v;
      }
    }
  }
}

// ------------------------------ instance norm ------------------------------
__global__ void inorm_stats_k(const float* __restrict__ x, float* __restrict__ st, int HW) {
  int bc = blockIdx.x;
  const float* p = x + (size_t)bc * HW;
  float s = 0.f, s2 = 0.f;
  for (int i = threadIdx.x; i < HW; i += 256) { float v = p[i]; s += v; s2 += v * v; }
  __shared__ float rs[256], rq[256];
  rs[threadIdx.x] = s; rq[threadIdx.x] = s2; __syncthreads();
  for (int o = 128; o > 0; o >>= 1) {
    if (threadIdx.x < o) { rs[threadIdx.x] += rs[threadIdx.x + o]; rq[threadIdx.x] += rq[threadIdx.x + o]; }
    __syncthreads();
  }
  if (threadIdx.x == 0) {
    float m = rs[0] / HW; float var = rq[0] / HW - m * m;
    st[2 * bc] = m; st[2 * bc + 1] = rsqrtf(var + 1e-5f);
  }
}

__global__ void inorm_apply_k(const float* __restrict__ x, const float* __restrict__ st,
                              const float* __restrict__ g, const float* __restrict__ b,
                              float* __restrict__ y, int C, int HW, size_t total) {
  EWIDX;
  int bc = (int)(i / (size_t)HW); int c = bc % C;
  float m = st[2 * bc], r = st[2 * bc + 1];
  y[i] = lrelu_f(g[c] * (x[i] - m) * r + b[c]);
}

// ------------------------------ elementwise --------------------------------
__global__ void copy_ch_k(const float* __restrict__ s, float* __restrict__ d,
                          int Cs, int Cd, int c0, size_t HW, size_t total) {
  EWIDX;
  size_t hw = i % HW; size_t t = i / HW; size_t c = t % Cs; size_t b = t / Cs;
  d[(b * Cd + c0 + c) * HW + hw] = s[i];
}
__global__ void diffx_k(const float* __restrict__ x, float* __restrict__ y, int W, size_t total) {
  EWIDX; int w = (int)(i % W); y[i] = (w + 1 < W) ? x[i + 1] - x[i] : 0.f;
}
__global__ void diffy_k(const float* __restrict__ x, float* __restrict__ y, int H, int W, size_t total) {
  EWIDX; int h = (int)((i / W) % H); y[i] = (h + 1 < H) ? x[i + W] - x[i] : 0.f;
}
__global__ void add3_lrelu_k(const float* __restrict__ a, const float* __restrict__ b,
                             const float* __restrict__ c, float* __restrict__ y, size_t total) {
  EWIDX; y[i] = lrelu_f(a[i] + b[i] + c[i]);
}
__global__ void xht_k(const float* __restrict__ x, float* __restrict__ y, int W, float sc, size_t total) {
  EWIDX; int w = (int)(i % W);
  float v; if (w == 0) v = -x[i + 1]; else if (w == W - 1) v = x[i]; else v = x[i] - x[i + 1];
  y[i] = sc * v;
}
__global__ void vtx_k(const float* __restrict__ x, float* __restrict__ y, int H, int W, float sc, size_t total) {
  EWIDX; int h = (int)((i / W) % H);
  float v; if (h == 0) v = -x[i + W]; else if (h == H - 1) v = x[i]; else v = x[i] - x[i + W];
  y[i] = sc * v;
}
__global__ void rhs1_k(const float* __restrict__ X, const float* __restrict__ Gt,
                       const float* __restrict__ P, const float* __restrict__ L,
                       float* __restrict__ y, float g, size_t total) {
  EWIDX; y[i] = X[i] + Gt[i] + g * P[i] + L[i];
}
__global__ void rhs2_k(const float* __restrict__ Gt, const float* __restrict__ X,
                       const float* __restrict__ L, float* __restrict__ y, float g, size_t total) {
  EWIDX; y[i] = Gt[i] + g * X[i] - L[i];
}
__global__ void lam_k(float* __restrict__ L, const float* __restrict__ P,
                      const float* __restrict__ X, size_t total) {
  EWIDX; L[i] += P[i] - X[i];
}
__global__ void zero_k(float* __restrict__ p, size_t total) { EWIDX; p[i] = 0.f; }

__global__ void avgpool2_k(const float* __restrict__ x, float* __restrict__ y,
                           int H, int W, size_t total) {
  EWIDX;
  int Wo = W >> 1, Ho = H >> 1;
  int wo = (int)(i % Wo); size_t t = i / Wo; int ho = (int)(t % Ho); size_t bc = t / Ho;
  const float* p = x + ((size_t)bc * H + 2 * ho) * W + 2 * wo;
  y[i] = 0.25f * (p[0] + p[1] + p[W] + p[W + 1]);
}
__global__ void up2_k(const float* __restrict__ x, float* __restrict__ y,
                      int H, int W, size_t total) {
  EWIDX;
  int Ho = 2 * H, Wo = 2 * W;
  int wo = (int)(i % Wo); size_t t = i / Wo; int ho = (int)(t % Ho); size_t bc = t / Ho;
  float py = ho * (float)(H - 1) / (float)(Ho - 1);
  float px = wo * (float)(W - 1) / (float)(Wo - 1);
  int y0 = (int)floorf(py), x0i = (int)floorf(px);
  int y1 = min(y0 + 1, H - 1), x1 = min(x0i + 1, W - 1);
  float wy = py - y0, wx = px - x0i;
  const float* p = x + (size_t)bc * H * W;
  y[i] = (1.f - wy) * ((1.f - wx) * p[y0 * W + x0i] + wx * p[y0 * W + x1])
       +        wy  * ((1.f - wx) * p[y1 * W + x0i] + wx * p[y1 * W + x1]);
}

// --------------------- mpom: tridiagonal build + Thomas ---------------------
__device__ __forceinline__ float Hm_(int i, int j) {
  if (j == i + 1) return -1.f; if (i == j && i > 0) return 1.f; return 0.f;
}
__device__ __forceinline__ float HHT_(int i, int j, int n) {
  if (j == 0) return -Hm_(i, 1);
  if (j == n - 1) return Hm_(i, n - 1);
  return Hm_(i, j) - Hm_(i, j + 1);
}
__device__ __forceinline__ float Vm_(int i, int j) {
  if (i == j && i > 0) return 1.f; if (j == i - 1) return -1.f; return 0.f;
}
__device__ __forceinline__ float VTV_(int i, int j, int n) {
  if (i == 0) return -Vm_(1, j);
  if (i == n - 1) return Vm_(n - 1, j);
  return Vm_(i, j) - Vm_(i + 1, j);
}
__global__ void tri_build_k(float* dl, float* dm, float* du, int n, float s, float ci, int isH) {
  int i = blockIdx.x * blockDim.x + threadIdx.x; if (i >= n) return;
  float a = (i > 0)     ? ci * (isH ? HHT_(i, i - 1, n) : VTV_(i, i - 1, n)) : 0.f;
  float b = s +           ci * (isH ? HHT_(i, i, n)     : VTV_(i, i, n));
  float c = (i < n - 1) ? ci * (isH ? HHT_(i, i + 1, n) : VTV_(i, i + 1, n)) : 0.f;
  dl[i] = a; dm[i] = b; du[i] = c;
}

// X @ inv(A): solve A^T z = y per contiguous row (lower=du[j-1], upper=dl[j+1])
__global__ void thomas_rows_k(const float* __restrict__ rhs, float* __restrict__ out,
                              const float* __restrict__ dl, const float* __restrict__ dm,
                              const float* __restrict__ du, int n, int nsys, int clip) {
  int s = blockIdx.x * blockDim.x + threadIdx.x;
  if (s >= nsys || n > 384) return;
  const float* yv = rhs + (size_t)s * n;
  float* z = out + (size_t)s * n;
  float cp[384], dp[384];
  float dia = dm[0], up = (n > 1) ? dl[1] : 0.f, inv = 1.f / dia;
  cp[0] = up * inv; dp[0] = yv[0] * inv;
  for (int j = 1; j < n; ++j) {
    float low = du[j - 1];
    dia = dm[j]; up = (j < n - 1) ? dl[j + 1] : 0.f;
    float m = 1.f / (dia - low * cp[j - 1]);
    cp[j] = up * m; dp[j] = (yv[j] - low * dp[j - 1]) * m;
  }
  float prev = dp[n - 1];
  z[n - 1] = clip ? clip01(prev) : prev;
  for (int j = n - 2; j >= 0; --j) { prev = dp[j] - cp[j] * prev; z[j] = clip ? clip01(prev) : prev; }
}

// inv(A) @ Y: solve A z = y per column (stride W)
__global__ void thomas_cols_k(const float* __restrict__ rhs, float* __restrict__ out,
                              const float* __restrict__ dl, const float* __restrict__ dm,
                              const float* __restrict__ du, int n, int W, int nsys, int clip) {
  int s = blockIdx.x * blockDim.x + threadIdx.x;
  if (s >= nsys || n > 384) return;
  int img = s / W, col = s % W;
  const float* yv = rhs + (size_t)img * n * W + col;
  float* z = out + (size_t)img * n * W + col;
  float cp[384], dp[384];
  float dia = dm[0], up = (n > 1) ? du[0] : 0.f, inv = 1.f / dia;
  cp[0] = up * inv; dp[0] = yv[0] * inv;
  for (int j = 1; j < n; ++j) {
    float low = dl[j];
    dia = dm[j]; up = (j < n - 1) ? du[j] : 0.f;
    float m = 1.f / (dia - low * cp[j - 1]);
    cp[j] = up * m; dp[j] = (yv[(size_t)j * W] - low * dp[j - 1]) * m;
  }
  float prev = dp[n - 1];
  z[(size_t)(n - 1) * W] = clip ? clip01(prev) : prev;
  for (int j = n - 2; j >= 0; --j) { prev = dp[j] - cp[j] * prev; z[(size_t)j * W] = clip ? clip01(prev) : prev; }
}

// ============================ host orchestration ============================
struct Arena {
  char* base; size_t off;
  float* alloc(size_t nfloats) {
    float* p = (float*)(base + off);
    off += ((nfloats * sizeof(float) + 255) & ~(size_t)255);
    return p;
  }
};
struct CBp  { const float *b1, *b2, *g1, *g2, *w1, *w2; };
struct Netp { CBp down, in1, in2, up; const float* neck; };
struct Gemp { const float *NX[3], *NY[3], *Rw1[3], *Rw2[3]; };

static inline unsigned ewb(size_t t) { return (unsigned)((t + 255) / 256); }

// pack weights + launch conv; pack buffer released immediately (stream order
// guarantees the next writer of that region runs after this conv finishes).
static void conv(hipStream_t st, Arena& A, const float* x, const float* w, const float* add,
                 float* y, int Bn, int Cin, int H, int W, int Cout, int taps3, int act) {
  int tpc = taps3 ? 9 : 1;
  int Kreal = Cin * tpc;
  int nCh = (Kreal + 31) >> 5;
  int nPair = (Cout + 31) >> 5;
  size_t halves = (size_t)(nPair * 2) * nCh * 512;
  size_t mk = A.off;
  _Float16* wp = (_Float16*)A.alloc((halves + 1) / 2);
  pack_w_k<<<ewb(halves), 256, 0, st>>>(w, wp, Cout, Cin, tpc, nCh, halves);
  dim3 g(W >> 4, H, Bn * nPair);
  size_t hdr = (size_t)(18 + (taps3 ? Cin * 54 : Cin << 4) + (nCh << 5)) * sizeof(_Float16);
  size_t sh = ((hdr + 15) & ~(size_t)15) + 4096;   // +16B-aligned async B staging
  conv_wmma_k<<<g, 32, sh, st>>>(x, (const v16h*)wp, add, y, Bn, Cin, H, W, Cout, taps3, act, nCh);
  A.off = mk;
}
static void copy_ch(hipStream_t st, const float* s, float* d, int Bn, int Cs, int Cd, int c0, size_t HW) {
  size_t total = (size_t)Bn * Cs * HW;
  copy_ch_k<<<ewb(total), 256, 0, st>>>(s, d, Cs, Cd, c0, HW, total);
}
static void inorm(hipStream_t st, float* x, float* stats, const float* g, const float* b,
                  int Bn, int C, size_t HW) {
  inorm_stats_k<<<Bn * C, 256, 0, st>>>(x, stats, (int)HW);
  size_t total = (size_t)Bn * C * HW;
  inorm_apply_k<<<ewb(total), 256, 0, st>>>(x, stats, g, b, x, C, (int)HW, total);
}

static float* conv_block_run(hipStream_t st, Arena& A, const float* x,
                             int Bn, int Cin, int H, int W, int outc, const CBp& cb) {
  int mid = Cin * 3; size_t HW = (size_t)H * W;
  float* out = A.alloc((size_t)Bn * outc * HW);
  size_t mk = A.off;
  float* t = A.alloc((size_t)Bn * mid * HW);
  float* stats = A.alloc((size_t)Bn * mid * 2 + 16);
  conv(st, A, x, cb.w1, nullptr, t, Bn, Cin, H, W, mid, 1, AC_LIN);
  inorm(st, t, stats, cb.g1, cb.b1, Bn, mid, HW);
  conv(st, A, t, cb.w2, nullptr, out, Bn, mid, H, W, outc, 1, AC_LIN);
  inorm(st, out, stats, cb.g2, cb.b2, Bn, outc, HW);
  A.off = mk;
  return out;
}

// gem(p, R, L) with RL = cat([Lp(cL), Rp(cR)]); outputs written to Sx, Sy, Ls.
static void gem_run(hipStream_t st, Arena& A, const Gemp& gp,
                    const float* Lp, int cL, const float* Rp, int cR, int outc,
                    float* Sx, float* Sy, float* Ls) {
  const int Bn = 2, H = 384, W = 384;
  const size_t HW = (size_t)H * W, U = (size_t)Bn * HW;
  float* RL4 = A.alloc(4 * U);
  copy_ch(st, Lp, RL4, Bn, cL, 4, 0, HW);
  copy_ch(st, Rp, RL4, Bn, cR, 4, cL, HW);
  float* NX = A.alloc(4 * U); diffx_k<<<ewb(4 * U), 256, 0, st>>>(RL4, NX, W, 4 * U);
  float* NY = A.alloc(4 * U); diffy_k<<<ewb(4 * U), 256, 0, st>>>(RL4, NY, H, W, 4 * U);
  float* t32 = A.alloc(32 * U);
  float* RL32 = A.alloc(32 * U);
  float* d32 = A.alloc(32 * U);
  float* s32 = A.alloc(32 * U);
  // block 1
  conv(st, A, RL4, gp.Rw1[0], nullptr, t32, Bn, 4, H, W, 32, 0, AC_LRELU);
  conv(st, A, t32, gp.Rw2[0], nullptr, RL32, Bn, 32, H, W, 32, 1, AC_LIN);
  float* NX32 = A.alloc(32 * U);
  diffx_k<<<ewb(32 * U), 256, 0, st>>>(RL32, d32, W, 32 * U);
  conv(st, A, NX, gp.NX[0], d32, NX32, Bn, 4, H, W, 32, 1, AC_TANH);
  float* NY32 = A.alloc(32 * U);
  diffy_k<<<ewb(32 * U), 256, 0, st>>>(RL32, d32, H, W, 32 * U);
  conv(st, A, NY, gp.NY[0], d32, NY32, Bn, 4, H, W, 32, 1, AC_TANH);
  // block 2
  add3_lrelu_k<<<ewb(32 * U), 256, 0, st>>>(RL32, NX32, NY32, s32, 32 * U);
  conv(st, A, s32, gp.Rw1[1], nullptr, t32, Bn, 32, H, W, 32, 0, AC_LRELU);
  conv(st, A, t32, gp.Rw2[1], nullptr, RL32, Bn, 32, H, W, 32, 1, AC_LIN);
  float* NX32b = A.alloc(32 * U);
  diffx_k<<<ewb(32 * U), 256, 0, st>>>(RL32, d32, W, 32 * U);
  conv(st, A, NX32, gp.NX[1], d32, NX32b, Bn, 32, H, W, 32, 1, AC_TANH);
  float* NY32b = A.alloc(32 * U);
  diffy_k<<<ewb(32 * U), 256, 0, st>>>(RL32, d32, H, W, 32 * U);
  conv(st, A, NY32, gp.NY[1], d32, NY32b, Bn, 32, H, W, 32, 1, AC_TANH);
  // block 3
  add3_lrelu_k<<<ewb(32 * U), 256, 0, st>>>(RL32, NX32b, NY32b, s32, 32 * U);
  conv(st, A, s32, gp.Rw1[2], nullptr, t32, Bn, 32, H, W, 32, 0, AC_LRELU);
  float* RLo = A.alloc((size_t)outc * U);
  conv(st, A, t32, gp.Rw2[2], nullptr, RLo, Bn, 32, H, W, outc, 1, AC_LIN);
  float* doc = A.alloc((size_t)outc * U);
  size_t to = (size_t)outc * U;
  diffx_k<<<ewb(to), 256, 0, st>>>(RLo, doc, W, to);
  conv(st, A, NX32b, gp.NX[2], doc, Sx, Bn, 32, H, W, outc, 1, AC_TANH);
  diffy_k<<<ewb(to), 256, 0, st>>>(RLo, doc, H, W, to);
  conv(st, A, NY32b, gp.NY[2], doc, Sy, Bn, 32, H, W, outc, 1, AC_TANH);
  add3_lrelu_k<<<ewb(to), 256, 0, st>>>(RLo, Sx, Sy, Ls, to);
}

static void mpom_run(hipStream_t st, Arena& A, const float* X0, const float* Gx, const float* Gy,
                     int C, float cd, float cinv, float g, float* P, float* dH, float* dV) {
  const int Bn = 2, H = 384, W = 384, n = 384;
  const size_t tot = (size_t)Bn * C * H * W;
  size_t base = A.off;
  tri_build_k<<<3, 128, 0, st>>>(dH, dH + n, dH + 2 * n, n, 1.f + g, cinv, 1);
  tri_build_k<<<3, 128, 0, st>>>(dV, dV + n, dV + 2 * n, n, g, cinv, 0);
  float* GxT = A.alloc(tot); xht_k<<<ewb(tot), 256, 0, st>>>(Gx, GxT, W, cd, tot);
  float* GyT = A.alloc(tot); vtx_k<<<ewb(tot), 256, 0, st>>>(Gy, GyT, H, W, cd, tot);
  float* X = A.alloc(tot);
  (void)hipMemcpyAsync(X, X0, tot * sizeof(float), hipMemcpyDeviceToDevice, st);
  (void)hipMemcpyAsync(P, X0, tot * sizeof(float), hipMemcpyDeviceToDevice, st);
  float* Lam = A.alloc(tot); zero_k<<<ewb(tot), 256, 0, st>>>(Lam, tot);
  float* rhs = A.alloc(tot);
  int nsysR = Bn * C * H, nsysC = Bn * C * W;
  for (int t = 0; t < 3; ++t) {
    rhs1_k<<<ewb(tot), 256, 0, st>>>(X, GxT, P, Lam, rhs, g, tot);
    thomas_rows_k<<<(nsysR + 63) / 64, 64, 0, st>>>(rhs, X, dH, dH + n, dH + 2 * n, n, nsysR, 1);
    rhs2_k<<<ewb(tot), 256, 0, st>>>(GyT, X, Lam, rhs, g, tot);
    thomas_cols_k<<<(nsysC + 63) / 64, 64, 0, st>>>(rhs, P, dV, dV + n, dV + 2 * n, n, W, nsysC, 1);
    lam_k<<<ewb(tot), 256, 0, st>>>(Lam, P, X, tot);
  }
  A.off = base;
}

static void unet_run(hipStream_t st, Arena& A, const Netp& np,
                     const float* Aa, const float* Bb, const float* Cc, int outc, float* out) {
  const int Bn = 2, H = 384, W = 384;
  const size_t HW = (size_t)H * W, U = (size_t)Bn * HW;
  size_t base = A.off;
  int inc0 = 3 * outc;
  float* cat = A.alloc((size_t)inc0 * U);
  copy_ch(st, Aa, cat, Bn, outc, inc0, 0, HW);
  copy_ch(st, Bb, cat, Bn, outc, inc0, outc, HW);
  copy_ch(st, Cc, cat, Bn, outc, inc0, 2 * outc, HW);
  float* dwn = conv_block_run(st, A, cat, Bn, inc0, H, W, 32, np.down);
  int H2 = H / 2, W2 = W / 2; size_t HW2 = (size_t)H2 * W2;
  float* pool = A.alloc((size_t)Bn * 32 * HW2);
  avgpool2_k<<<ewb((size_t)Bn * 32 * HW2), 256, 0, st>>>(dwn, pool, H, W, (size_t)Bn * 32 * HW2);
  float* x1 = conv_block_run(st, A, pool, Bn, 32, H2, W2, 32, np.in1);
  float* x2 = conv_block_run(st, A, x1, Bn, 32, H2, W2, 32, np.in2);
  float* up = A.alloc(32 * U);
  up2_k<<<ewb(32 * U), 256, 0, st>>>(x2, up, H2, W2, 32 * U);
  int cin2 = inc0 + 32;
  float* cat2 = A.alloc((size_t)cin2 * U);
  copy_ch(st, cat, cat2, Bn, inc0, cin2, 0, HW);
  copy_ch(st, up, cat2, Bn, 32, cin2, inc0, HW);
  float* xu = conv_block_run(st, A, cat2, Bn, cin2, H, W, 32, np.up);
  conv(st, A, xu, np.neck, Cc, out, Bn, 32, H, W, outc, 1, AC_LRELU);
  A.off = base;
}

static void lem_run(hipStream_t st, Arena& A, const Gemp& se, const Netp& nn,
                    const float* L, const float* R, float beta1, float* Lout,
                    float* dH, float* dV) {
  const size_t U = (size_t)2 * 384 * 384;
  size_t base = A.off;
  float* Sx = A.alloc(U); float* Sy = A.alloc(U); float* Ls = A.alloc(U);
  size_t mk = A.off;
  gem_run(st, A, se, /*cat=[R(3),L(1)]*/ R, 3, L, 1, 1, Sx, Sy, Ls);
  A.off = mk;
  float* Lm = A.alloc(U);
  mpom_run(st, A, L, Sx, Sy, 1, beta1, beta1 + 0.1f, 1.0f, Lm, dH, dV);
  unet_run(st, A, nn, Lm, Ls, L, 1, Lout);
  A.off = base;
}
static void rem_run(hipStream_t st, Arena& A, const Gemp& we, const Netp& nn,
                    const float* R, const float* L, float alpha, float* Rout,
                    float* dH, float* dV) {
  const size_t U = (size_t)2 * 384 * 384;
  size_t base = A.off;
  float* Sx = A.alloc(3 * U); float* Sy = A.alloc(3 * U); float* Ls = A.alloc(3 * U);
  size_t mk = A.off;
  gem_run(st, A, we, /*cat=[L(1),R(3)]*/ L, 1, R, 3, 3, Sx, Sy, Ls);
  A.off = mk;
  float* Rm = A.alloc(3 * U);
  mpom_run(st, A, R, Sx, Sy, 3, alpha, alpha, 1.0f, Rm, dH, dV);
  unet_run(st, A, nn, Rm, Ls, R, 3, Rout);
  A.off = base;
}

// ------- param decoding: assumes jax tree (sorted-key) flattening ----------
static const float* P_(void* const* d, int n, int& cur) {
  int i = (cur < n) ? cur : (n > 0 ? n - 1 : 0); ++cur;
  return (const float*)d[i];
}
static CBp readCB(void* const* d, int n, int& c) {
  CBp r; r.b1 = P_(d, n, c); r.b2 = P_(d, n, c); r.g1 = P_(d, n, c);
  r.g2 = P_(d, n, c); r.w1 = P_(d, n, c); r.w2 = P_(d, n, c); return r;
}
static Netp readNet(void* const* d, int n, int& c) {  // sorted: down,in1,in2,neck,up
  Netp r; r.down = readCB(d, n, c); r.in1 = readCB(d, n, c); r.in2 = readCB(d, n, c);
  r.neck = P_(d, n, c); r.up = readCB(d, n, c); return r;
}
static Gemp readGem(void* const* d, int n, int& c) {  // NXB1-3,NYB1-3,RLB1-3{w1,w2}
  Gemp r;
  for (int k = 0; k < 3; ++k) r.NX[k] = P_(d, n, c);
  for (int k = 0; k < 3; ++k) r.NY[k] = P_(d, n, c);
  for (int k = 0; k < 3; ++k) { r.Rw1[k] = P_(d, n, c); r.Rw2[k] = P_(d, n, c); }
  return r;
}

extern "C" void kernel_launch(void* const* d_in, const int* in_sizes, int n_in,
                              void* d_out, int out_size, void* d_ws, size_t ws_size,
                              hipStream_t stream) {
  (void)in_sizes; (void)out_size; (void)ws_size;
  const float* LLp = (const float*)d_in[0];
  const float* RLp = (const float*)d_in[1];
  int cur = 2;
  Netp LEM[3], REM[3]; Gemp SE[3], WE[3];
  for (int k = 0; k < 3; ++k) LEM[k] = readNet(d_in, n_in, cur);
  for (int k = 0; k < 3; ++k) REM[k] = readNet(d_in, n_in, cur);
  for (int k = 0; k < 3; ++k) SE[k] = readGem(d_in, n_in, cur);
  for (int k = 0; k < 3; ++k) WE[k] = readGem(d_in, n_in, cur);

  Arena A{(char*)d_ws, 0};
  float* dH = A.alloc(3 * 384);
  float* dV = A.alloc(3 * 384);
  const size_t U = (size_t)2 * 384 * 384;
  float* L1 = A.alloc(U);     float* R1 = A.alloc(3 * U);
  float* L2 = A.alloc(U);     float* R2 = A.alloc(3 * U);
  float* outp = (float*)d_out;
  float* L3 = outp;           float* R3 = outp + U;

  lem_run(stream, A, SE[0], LEM[0], LLp, RLp, 25.f, L1, dH, dV);
  rem_run(stream, A, WE[0], REM[0], RLp, L1, 25.f, R1, dH, dV);
  lem_run(stream, A, SE[1], LEM[1], L1, R1, 50.f, L2, dH, dV);
  rem_run(stream, A, WE[1], REM[1], R1, L2, 50.f, R2, dH, dV);
  lem_run(stream, A, SE[2], LEM[2], L2, R2, 100.f, L3, dH, dV);
  rem_run(stream, A, WE[2], REM[2], R2, L3, 100.f, R3, dH, dV);
}
